// Mamba2Block_34926674051441
// MI455X (gfx1250) — compile-verified, matrix operand order checked by reading, not run on hardware
//
#include <hip/hip_runtime.h>
#include <hip/hip_bf16.h>

typedef __attribute__((ext_vector_type(16))) _Float16 v16h;
typedef __attribute__((ext_vector_type(8)))  _Float16 v8h;
typedef __attribute__((ext_vector_type(8)))  float    v8f;
typedef __attribute__((ext_vector_type(4)))  unsigned u32x4;
typedef __attribute__((ext_vector_type(8)))  int      i32x8;
typedef __attribute__((ext_vector_type(4)))  int      i32x4;

#define DM        1024
#define DI        2048
#define DS        128
#define NH        32
#define HD        64
#define CONVD     2304      // DI + 2*DS
#define NPROJ     4384      // 2*DI + 2*DS + NH
#define NPROJ_PAD 4416      // round up to 64
#define SEQ       1024
#define TOK       2048      // BATCH * SEQLEN
#define NINTER    2752

__device__ __forceinline__ float silu_f(float x) { return x / (1.0f + __expf(-x)); }

// ---------------------------------------------------------------------------
// f32 -> f16 weight conversion with optional row padding (rows n >= Nreal -> 0)
// ---------------------------------------------------------------------------
__global__ void cvt_f16_pad(const float* __restrict__ src, _Float16* __restrict__ dst,
                            int Nreal, int Npad, int K) {
    size_t idx = (size_t)blockIdx.x * blockDim.x + threadIdx.x;
    size_t total = (size_t)Npad * K;
    if (idx >= total) return;
    int n = (int)(idx / K);
    int k = (int)(idx - (size_t)n * K);
    dst[idx] = (n < Nreal) ? (_Float16)src[(size_t)n * K + k] : (_Float16)0.0f;
}

// ---------------------------------------------------------------------------
// Row RMS-norm (f32 in, f16 out): out = x * rsqrt(mean(x^2)+eps) * w
// ---------------------------------------------------------------------------
template<int KD>
__global__ void rmsnorm_to_f16(const float* __restrict__ x, const float* __restrict__ w,
                               _Float16* __restrict__ out) {
    int row = blockIdx.x;
    const float* xr = x + (size_t)row * KD;
    float ss = 0.0f;
    for (int i = threadIdx.x; i < KD; i += blockDim.x) { float v = xr[i]; ss += v * v; }
#pragma unroll
    for (int off = 16; off > 0; off >>= 1) ss += __shfl_xor(ss, off, 32);
    __shared__ float red[8];
    if ((threadIdx.x & 31) == 0) red[threadIdx.x >> 5] = ss;
    __syncthreads();
    float tot = 0.0f;
#pragma unroll
    for (int i = 0; i < 8; ++i) tot += red[i];
    float scale = rsqrtf(tot / (float)KD + 1e-6f);
    for (int i = threadIdx.x; i < KD; i += blockDim.x)
        out[(size_t)row * KD + i] = (_Float16)(xr[i] * scale * w[i]);
}

// ---------------------------------------------------------------------------
// Gated RMS-norm for SSM output: v = y * silu(z); out = rmsnorm(v) * w  (f16)
// ---------------------------------------------------------------------------
__global__ void gate_rmsnorm_to_f16(const float* __restrict__ y, const float* __restrict__ zx,
                                    const float* __restrict__ w, _Float16* __restrict__ out) {
    int row = blockIdx.x;
    const float* yr = y + (size_t)row * DI;
    const float* zr = zx + (size_t)row * NPROJ_PAD;
    float ss = 0.0f;
    for (int i = threadIdx.x; i < DI; i += blockDim.x) {
        float v = yr[i] * silu_f(zr[i]);
        ss += v * v;
    }
#pragma unroll
    for (int off = 16; off > 0; off >>= 1) ss += __shfl_xor(ss, off, 32);
    __shared__ float red[8];
    if ((threadIdx.x & 31) == 0) red[threadIdx.x >> 5] = ss;
    __syncthreads();
    float tot = 0.0f;
#pragma unroll
    for (int i = 0; i < 8; ++i) tot += red[i];
    float scale = rsqrtf(tot / (float)DI + 1e-6f);
    for (int i = threadIdx.x; i < DI; i += blockDim.x) {
        float v = yr[i] * silu_f(zr[i]);
        out[(size_t)row * DI + i] = (_Float16)(v * scale * w[i]);
    }
}

// ---------------------------------------------------------------------------
// Causal depthwise conv (width 4) + bias + silu over xBC columns of zxbcdt.
// ---------------------------------------------------------------------------
__global__ void conv_silu(const float* __restrict__ zx, const float* __restrict__ cw,
                          const float* __restrict__ cb, float* __restrict__ xc) {
    size_t idx = (size_t)blockIdx.x * blockDim.x + threadIdx.x;
    if (idx >= (size_t)TOK * CONVD) return;
    int c = (int)(idx % CONVD);
    int t = (int)(idx / CONVD);
    int l = t % SEQ;
    float s = cb[c];
#pragma unroll
    for (int k = 0; k < 4; ++k) {
        int ll = l - 3 + k;
        if (ll >= 0)
            s += zx[(size_t)(t - 3 + k) * NPROJ_PAD + DI + c] * cw[c * 4 + k];
    }
    xc[idx] = silu_f(s);
}

// ---------------------------------------------------------------------------
// dt' = softplus(dt + dt_bias)
// ---------------------------------------------------------------------------
__global__ void dt_softplus(const float* __restrict__ zx, const float* __restrict__ dtb,
                            float* __restrict__ dt) {
    int idx = blockIdx.x * blockDim.x + threadIdx.x;
    if (idx >= TOK * NH) return;
    int h = idx % NH;
    int t = idx / NH;
    float v = zx[(size_t)t * NPROJ_PAD + (NPROJ - NH) + h] + dtb[h];
    dt[idx] = (v > 20.0f) ? v : log1pf(__expf(v));
}

// ---------------------------------------------------------------------------
// Sequential SSM scan. One block per (head, batch); 64x128 state in registers.
// ---------------------------------------------------------------------------
__global__ void ssm_scan(const float* __restrict__ xc, const float* __restrict__ dt,
                         const float* __restrict__ A_log, const float* __restrict__ Dp,
                         float* __restrict__ yb) {
    int head = blockIdx.x;
    int b = blockIdx.y;
    int tid = threadIdx.x;
    int p = tid >> 2;
    int q = tid & 3;
    float Ah = -__expf(A_log[head]);
    float Dh = Dp[head];
    float h[32];
#pragma unroll
    for (int i = 0; i < 32; ++i) h[i] = 0.0f;
    __shared__ float sB[DS];
    __shared__ float sC[DS];
    for (int l = 0; l < SEQ; ++l) {
        int tok = b * SEQ + l;
        size_t toff = (size_t)tok * CONVD;
        __syncthreads();
        if (tid < DS)            sB[tid]        = xc[toff + DI + tid];
        else if (tid < 2 * DS)   sC[tid - DS]   = xc[toff + DI + DS + (tid - DS)];
        __syncthreads();
        float dtv  = dt[tok * NH + head];
        float xv   = xc[toff + head * HD + p];
        float dA   = __expf(dtv * Ah);
        float coef = dtv * xv;
        float y = 0.0f;
        const float* Bq = sB + q * 32;
        const float* Cq = sC + q * 32;
#pragma unroll
        for (int i = 0; i < 32; ++i) {
            h[i] = h[i] * dA + coef * Bq[i];
            y += h[i] * Cq[i];
        }
        y += __shfl_xor(y, 1, 32);
        y += __shfl_xor(y, 2, 32);
        if (q == 0)
            yb[(size_t)tok * DI + head * HD + p] = y + Dh * xv;
    }
}

// ---------------------------------------------------------------------------
// TDM: issue a 2-D tile DMA (tile_rows x 32 f16 elements) Global -> LDS.
// D# per CDNA5 ISA ch.8: group0 = {count, lds_addr, global_addr, type=2},
// group1 = {data_size=2B, tensor_dim0=K, tensor_dim1=rows, tile_dim0=32,
// tile_dim1=tile_rows, tensor_dim0_stride=K}. Groups 2/3 zero (2-D tile).
// This toolchain exposes the 6-arg builtin (clang-23 / therock-10.0).
// ---------------------------------------------------------------------------
__device__ __forceinline__ unsigned lds_off_of(const void* p) {
    return (unsigned)(unsigned long long)(uintptr_t)p;   // flat addr low 32 = LDS offset
}

__device__ __forceinline__ void tdm_load_tile(const _Float16* gbase, unsigned lds_off,
                                              int Krow, int rows_total, int row0, int k0,
                                              int tile_rows) {
    unsigned long long ga = (unsigned long long)(uintptr_t)gbase
                          + ((unsigned long long)(unsigned)row0 * (unsigned)Krow
                             + (unsigned)k0) * 2ull;
    u32x4 g0;
    g0[0] = 1u;                                                   // count=1 (user D#)
    g0[1] = lds_off;                                              // lds_addr
    g0[2] = (unsigned)ga;                                         // global_addr[31:0]
    g0[3] = (unsigned)((ga >> 32) & 0x1FFFFFFull) | (2u << 30);   // ga[56:32], type=2
    unsigned K = (unsigned)Krow, R = (unsigned)rows_total;
    i32x8 g1;
    g1[0] = (int)(1u << 16);                                      // data_size=1 -> 2 bytes
    g1[1] = (int)((K & 0xFFFFu) << 16);                           // tensor_dim0 lo
    g1[2] = (int)(((K >> 16) & 0xFFFFu) | ((R & 0xFFFFu) << 16)); // dim0 hi | dim1 lo
    g1[3] = (int)(((R >> 16) & 0xFFFFu) | (32u << 16));           // dim1 hi | tile_dim0=32
    g1[4] = (int)((unsigned)tile_rows & 0xFFFFu);                 // tile_dim1 | tile_dim2=0
    g1[5] = (int)K;                                               // tensor_dim0_stride lo
    g1[6] = 0;                                                    // stride hi | dim1_stride
    g1[7] = 0;
    i32x4 z4 = {0, 0, 0, 0};
    i32x8 z8 = {0, 0, 0, 0, 0, 0, 0, 0};
    __builtin_amdgcn_tensor_load_to_lds(g0, g1, z4, z4, z8, 0);
}

// ---------------------------------------------------------------------------
// 16x32 f16 fragment from an LDS tile (row-major, 32 halfs per row).
// ISA 16-bit A/B layout: lanes 0-15: K {0..7,16..23}; lanes 16-31: +8.
// ---------------------------------------------------------------------------
__device__ __forceinline__ v16h frag_from_lds(const _Float16* sbase, int row0) {
    int lane = threadIdx.x & 31;
    int r    = lane & 15;
    int hi   = lane >> 4;
    const _Float16* q = sbase + (size_t)(row0 + r) * 32 + hi * 8;
    v8h lo = *(const v8h*)(q);
    v8h hh = *(const v8h*)(q + 16);
    v16h out;
#pragma unroll
    for (int i = 0; i < 8; ++i) { out[i] = lo[i]; out[i + 8] = hh[i]; }
    return out;
}

// ---------------------------------------------------------------------------
// WMMA GEMM with TDM double-buffered LDS staging.
// C[M,N] = A[M,K] (f16 row-major) x W[N,K]^T (f16 row-major).
// Block = 256 threads (8 wave32), C-tile 128x64. Wave 0 drives the TDM:
// per K-step one 128x32 A-tile + one 64x32 B-tile, double buffered (24 KB LDS).
// EPI: 0 = store f32; 1 = store f32 + aux residual; 2 = f16 silu(aux)*acc.
// ---------------------------------------------------------------------------
template<int EPI>
__global__ void gemm_wmma_f16(const _Float16* __restrict__ A, const _Float16* __restrict__ W,
                              int M, int N, int K,
                              float* __restrict__ outF, _Float16* __restrict__ outH,
                              const float* __restrict__ aux) {
    __shared__ _Float16 sA[2][128 * 32];
    __shared__ _Float16 sB[2][64 * 32];

    int wave = threadIdx.x >> 5;
    int lane = threadIdx.x & 31;
    int mblk = blockIdx.y * 128;
    int m0 = mblk + wave * 16;
    int n0 = blockIdx.x * 64;

    v8f acc[4];
#pragma unroll
    for (int j = 0; j < 4; ++j)
#pragma unroll
        for (int i = 0; i < 8; ++i) acc[j][i] = 0.0f;

    int steps = K >> 5;
    if (wave == 0) {
        tdm_load_tile(A, lds_off_of(&sA[0][0]), K, M, mblk, 0, 128);
        tdm_load_tile(W, lds_off_of(&sB[0][0]), K, N, n0, 0, 64);
    }
    for (int i = 0; i < steps; ++i) {
        int cur = i & 1;
        if (wave == 0) {
            if (i + 1 < steps) {
                tdm_load_tile(A, lds_off_of(&sA[cur ^ 1][0]), K, M, mblk, (i + 1) * 32, 128);
                tdm_load_tile(W, lds_off_of(&sB[cur ^ 1][0]), K, N, n0, (i + 1) * 32, 64);
                __builtin_amdgcn_s_wait_tensorcnt(2);   // buffers for step i resident
            } else {
                __builtin_amdgcn_s_wait_tensorcnt(0);
            }
        }
        __syncthreads();
        v16h a = frag_from_lds(sA[cur], wave * 16);
#pragma unroll
        for (int j = 0; j < 4; ++j) {
            v16h bfrag = frag_from_lds(sB[cur], j * 16);
            acc[j] = __builtin_amdgcn_wmma_f32_16x16x32_f16(
                false, a, false, bfrag, (short)0, acc[j], false, false);
        }
        __syncthreads();
    }

    // C/D layout: lane n = lane&15; VGPR v holds M = 8*(lane>>4) + v.
    int nlane = lane & 15;
    int mbase = (lane >> 4) * 8;
#pragma unroll
    for (int j = 0; j < 4; ++j) {
        int n = n0 + j * 16 + nlane;
#pragma unroll
        for (int v = 0; v < 8; ++v) {
            int m = m0 + mbase + v;
            size_t idx = (size_t)m * N + n;
            float r = acc[j][v];
            if (EPI == 0) {
                outF[idx] = r;
            } else if (EPI == 1) {
                outF[idx] = r + aux[idx];
            } else {
                outH[idx] = (_Float16)(silu_f(aux[idx]) * r);
            }
        }
    }
}

// ---------------------------------------------------------------------------
// Host-side orchestration
// ---------------------------------------------------------------------------
extern "C" void kernel_launch(void* const* d_in, const int* in_sizes, int n_in,
                              void* d_out, int out_size, void* d_ws, size_t ws_size,
                              hipStream_t stream) {
    (void)in_sizes; (void)n_in; (void)out_size; (void)ws_size;
    const float* hidden      = (const float*)d_in[0];
    const float* norm_w      = (const float*)d_in[1];
    const float* in_proj_w   = (const float*)d_in[2];
    const float* conv_w      = (const float*)d_in[3];
    const float* conv_b      = (const float*)d_in[4];
    const float* dt_bias     = (const float*)d_in[5];
    const float* A_log       = (const float*)d_in[6];
    const float* Dp          = (const float*)d_in[7];
    const float* ssm_norm_w  = (const float*)d_in[8];
    const float* out_proj_w  = (const float*)d_in[9];
    const float* post_norm_w = (const float*)d_in[10];
    const float* gate_w      = (const float*)d_in[11];
    const float* up_w        = (const float*)d_in[12];
    const float* down_w      = (const float*)d_in[13];
    float* out = (float*)d_out;

    char* ws = (char*)d_ws;
    size_t off = 0;
    auto alloc = [&](size_t bytes) -> void* {
        void* p = ws + off;
        off = (off + bytes + 255) & ~(size_t)255;
        return p;
    };

    _Float16* wA = (_Float16*)alloc((size_t)NPROJ_PAD * DM * 2);   // in_proj (padded)
    _Float16* wO = (_Float16*)alloc((size_t)DM * DI * 2);          // out_proj
    _Float16* wG = (_Float16*)alloc((size_t)NINTER * DM * 2);      // gate
    _Float16* wU = (_Float16*)alloc((size_t)NINTER * DM * 2);      // up
    _Float16* wD = (_Float16*)alloc((size_t)DM * NINTER * 2);      // down
    _Float16* hn = (_Float16*)alloc((size_t)TOK * DM * 2);         // rmsnorm(hidden) f16
    float*    zx = (float*)   alloc((size_t)TOK * NPROJ_PAD * 4);  // zxbcdt (padded)
    float*    xc = (float*)   alloc((size_t)TOK * CONVD * 4);      // conv+silu output
    float*    dtb= (float*)   alloc((size_t)TOK * NH * 4);         // softplus dt
    float*    yb = (float*)   alloc((size_t)TOK * DI * 4);         // scan output
    _Float16* yn = (_Float16*)alloc((size_t)TOK * DI * 2);         // gated rmsnorm f16
    float*    xb = (float*)   alloc((size_t)TOK * DM * 4);         // residual after mixer
    _Float16* h2 = (_Float16*)alloc((size_t)TOK * DM * 2);         // post-norm f16
    float*    gb = (float*)   alloc((size_t)TOK * NINTER * 4);     // gate GEMM output
    _Float16* mb = (_Float16*)alloc((size_t)TOK * NINTER * 2);     // silu(g)*u f16

    dim3 blk(256);
    auto cdiv = [](size_t a, size_t b) { return (unsigned)((a + b - 1) / b); };

    // Weight conversions (f32 -> f16, in_proj padded to 4416 rows)
    cvt_f16_pad<<<cdiv((size_t)NPROJ_PAD * DM, 256), blk, 0, stream>>>(in_proj_w, wA, NPROJ, NPROJ_PAD, DM);
    cvt_f16_pad<<<cdiv((size_t)DM * DI, 256),        blk, 0, stream>>>(out_proj_w, wO, DM, DM, DI);
    cvt_f16_pad<<<cdiv((size_t)NINTER * DM, 256),    blk, 0, stream>>>(gate_w, wG, NINTER, NINTER, DM);
    cvt_f16_pad<<<cdiv((size_t)NINTER * DM, 256),    blk, 0, stream>>>(up_w, wU, NINTER, NINTER, DM);
    cvt_f16_pad<<<cdiv((size_t)DM * NINTER, 256),    blk, 0, stream>>>(down_w, wD, DM, DM, NINTER);

    // 1) pre-norm
    rmsnorm_to_f16<DM><<<TOK, blk, 0, stream>>>(hidden, norm_w, hn);
    // 2) in_proj GEMM: (2048 x 1024) x (4416 x 1024)^T -> zxbcdt
    gemm_wmma_f16<0><<<dim3(NPROJ_PAD / 64, TOK / 128), blk, 0, stream>>>(
        hn, wA, TOK, NPROJ_PAD, DM, zx, nullptr, nullptr);
    // 3) causal conv + silu on xBC
    conv_silu<<<cdiv((size_t)TOK * CONVD, 256), blk, 0, stream>>>(zx, conv_w, conv_b, xc);
    // 4) softplus(dt + bias)
    dt_softplus<<<cdiv((size_t)TOK * NH, 256), blk, 0, stream>>>(zx, dt_bias, dtb);
    // 5) sequential SSM scan, one block per (head, batch)
    ssm_scan<<<dim3(NH, 2), blk, 0, stream>>>(xc, dtb, A_log, Dp, yb);
    // 6) y * silu(z), rmsnorm -> f16
    gate_rmsnorm_to_f16<<<TOK, blk, 0, stream>>>(yb, zx, ssm_norm_w, yn);
    // 7) out_proj GEMM + residual(hidden) -> xb
    gemm_wmma_f16<1><<<dim3(DM / 64, TOK / 128), blk, 0, stream>>>(
        yn, wO, TOK, DM, DI, xb, nullptr, hidden);
    // 8) post-norm
    rmsnorm_to_f16<DM><<<TOK, blk, 0, stream>>>(xb, post_norm_w, h2);
    // 9) gate GEMM -> gb (f32)
    gemm_wmma_f16<0><<<dim3(NINTER / 64, TOK / 128), blk, 0, stream>>>(
        h2, wG, TOK, NINTER, DM, gb, nullptr, nullptr);
    // 10) up GEMM fused with silu(gate)*up -> f16
    gemm_wmma_f16<2><<<dim3(NINTER / 64, TOK / 128), blk, 0, stream>>>(
        h2, wU, TOK, NINTER, DM, nullptr, mb, gb);
    // 11) down GEMM + residual(xb) -> out
    gemm_wmma_f16<1><<<dim3(DM / 64, TOK / 128), blk, 0, stream>>>(
        mb, wD, TOK, DM, NINTER, out, nullptr, xb);
}
